// GVAE_72164040507472
// MI455X (gfx1250) — compile-verified, matrix operand order checked by reading, not run on hardware
//
#include <hip/hip_runtime.h>
#include <hip/hip_bf16.h>

// ---------------- static problem sizes (match reference) ----------------
constexpr int BG   = 64;            // graphs
constexpr int NPG  = 96;            // nodes per graph
constexpr int F_IN = 30;
constexpr int NN   = BG * NPG;      // 6144 nodes
constexpr int DEG  = 16;
constexpr int MM   = NN * DEG;      // 98304 edges
constexpr int HEADS = 3;
constexpr int D_LAT = 24;
constexpr int PP   = NPG * (NPG - 1);   // 9120 ordered pairs / graph
constexpr int RR   = BG * PP;           // 583680 decoder rows

typedef _Float16 v16h __attribute__((ext_vector_type(16)));
typedef _Float16 v8h  __attribute__((ext_vector_type(8)));
typedef float    v8f  __attribute__((ext_vector_type(8)));

// ---------------- helpers ----------------
__device__ __forceinline__ float atomicMaxFloat(float* addr, float val) {
    // classic signed/unsigned trick; valid for mixed signs, init with -inf
    if (val >= 0.0f)
        return __int_as_float(atomicMax((int*)addr, __float_as_int(val)));
    else
        return __uint_as_float(atomicMin((unsigned int*)addr, (unsigned int)__float_as_int(val)));
}

__device__ __forceinline__ float sigmoidf(float x) { return 1.0f / (1.0f + expf(-x)); }

// =======================================================================
// Encoder kernels (TransformerConv, small: plain VALU + atomics)
// =======================================================================
__global__ void gvae_qkv(const float* __restrict__ x, int n_, int din, int hd, int dout,
                         const float* __restrict__ Wq, const float* __restrict__ bq,
                         const float* __restrict__ Wk, const float* __restrict__ bk,
                         const float* __restrict__ Wv, const float* __restrict__ bv,
                         const float* __restrict__ Ws, const float* __restrict__ bs,
                         float* __restrict__ q, float* __restrict__ k,
                         float* __restrict__ v, float* __restrict__ xr) {
    int idx = blockIdx.x * blockDim.x + threadIdx.x;
    if (idx >= n_ * hd) return;
    int n = idx / hd, j = idx % hd;
    const float* xn = x + (size_t)n * din;
    float aq = bq[j], ak = bk[j], av = bv[j];
    for (int i = 0; i < din; ++i) {
        float xv = xn[i];
        aq += xv * Wq[i * hd + j];
        ak += xv * Wk[i * hd + j];
        av += xv * Wv[i * hd + j];
    }
    q[idx] = aq; k[idx] = ak; v[idx] = av;
    if (j < dout) {
        float a = bs[j];
        for (int i = 0; i < din; ++i) a += xn[i] * Ws[i * dout + j];
        xr[n * dout + j] = a;
    }
}

__global__ void gvae_attn_init(float* __restrict__ amax, float* __restrict__ den,
                               float* __restrict__ acc, int nh, int na) {
    int i = blockIdx.x * blockDim.x + threadIdx.x;
    if (i < nh) { amax[i] = -__builtin_inff(); den[i] = 0.0f; }
    if (i < na) acc[i] = 0.0f;
}

__global__ void gvae_alpha(const float* __restrict__ q, const float* __restrict__ k,
                           const float* __restrict__ ea, const float* __restrict__ We,
                           const int* __restrict__ ei, int m_, int dout, int hd,
                           float* __restrict__ alpha, float* __restrict__ amax) {
    int idx = blockIdx.x * blockDim.x + threadIdx.x;
    if (idx >= m_ * HEADS) return;
    int m = idx / HEADS, h = idx % HEADS;
    int src = ei[m], dst = ei[m_ + m];
    const float* qd  = q + (size_t)dst * hd + h * dout;
    const float* ks  = k + (size_t)src * hd + h * dout;
    float e0 = ea[m * 2], e1 = ea[m * 2 + 1];
    const float* We0 = We + h * dout;
    const float* We1 = We + hd + h * dout;
    float a = 0.0f;
    for (int c = 0; c < dout; ++c) {
        float ec = e0 * We0[c] + e1 * We1[c];
        a += qd[c] * (ks[c] + ec);
    }
    a *= rsqrtf((float)dout);
    alpha[idx] = a;
    atomicMaxFloat(&amax[dst * HEADS + h], a);
}

__global__ void gvae_exden(const int* __restrict__ ei, int m_,
                           const float* __restrict__ amax,
                           float* __restrict__ alpha, float* __restrict__ den) {
    int idx = blockIdx.x * blockDim.x + threadIdx.x;
    if (idx >= m_ * HEADS) return;
    int m = idx / HEADS, h = idx % HEADS;
    int dst = ei[m_ + m];
    float mx = amax[dst * HEADS + h];
    if (!(mx > -1e30f)) mx = 0.0f;               // guard isolated nodes
    float ex = expf(alpha[idx] - mx);
    alpha[idx] = ex;
    atomicAdd(&den[dst * HEADS + h], ex);
}

__global__ void gvae_scatter(const float* __restrict__ v, const float* __restrict__ ea,
                             const float* __restrict__ We, const int* __restrict__ ei,
                             int m_, int dout, int hd,
                             const float* __restrict__ alpha, const float* __restrict__ den,
                             float* __restrict__ acc) {
    int idx = blockIdx.x * blockDim.x + threadIdx.x;
    if (idx >= m_ * HEADS) return;
    int m = idx / HEADS, h = idx % HEADS;
    int src = ei[m], dst = ei[m_ + m];
    float d = den[dst * HEADS + h];
    float w = alpha[idx] / fmaxf(d, 1e-16f);
    float e0 = ea[m * 2], e1 = ea[m * 2 + 1];
    const float* We0 = We + h * dout;
    const float* We1 = We + hd + h * dout;
    const float* vs  = v + (size_t)src * hd + h * dout;
    float* ad = acc + (size_t)dst * hd + h * dout;
    for (int c = 0; c < dout; ++c) {
        float ec = e0 * We0[c] + e1 * We1[c];
        atomicAdd(&ad[c], (vs[c] + ec) * w);
    }
}

__global__ void gvae_combine(const float* __restrict__ acc, const float* __restrict__ xr,
                             const float* __restrict__ Wb, int n_, int dout, int hd,
                             int do_relu, float* __restrict__ hout) {
    int n = blockIdx.x * blockDim.x + threadIdx.x;
    if (n >= n_) return;
    float outc[48];
    float dotb = 0.0f;
    for (int c = 0; c < dout; ++c) {
        float o = (acc[(size_t)n * hd + c] + acc[(size_t)n * hd + dout + c] +
                   acc[(size_t)n * hd + 2 * dout + c]) * (1.0f / 3.0f);
        float xv = xr[n * dout + c];
        dotb += o * Wb[c] + xv * Wb[dout + c] + (o - xv) * Wb[2 * dout + c];
        outc[c] = o;
    }
    float beta = sigmoidf(dotb);
    for (int c = 0; c < dout; ++c) {
        float xv = xr[n * dout + c];
        float hv = beta * xv + (1.0f - beta) * outc[c];
        if (do_relu) hv = fmaxf(hv, 0.0f);
        hout[n * dout + c] = hv;
    }
}

// ---------------- BatchNorm helpers ----------------
__global__ void gvae_colstats_f32(const float* __restrict__ src, int rows, int cols,
                                  float* __restrict__ mean, float* __restrict__ var) {
    int c = blockIdx.x;
    __shared__ float ssum[256], ssq[256];
    float s = 0.0f, s2 = 0.0f;
    for (int r = threadIdx.x; r < rows; r += blockDim.x) {
        float v = src[(size_t)r * cols + c];
        s += v; s2 += v * v;
    }
    ssum[threadIdx.x] = s; ssq[threadIdx.x] = s2;
    __syncthreads();
    for (int st = 128; st; st >>= 1) {
        if ((int)threadIdx.x < st) {
            ssum[threadIdx.x] += ssum[threadIdx.x + st];
            ssq[threadIdx.x]  += ssq[threadIdx.x + st];
        }
        __syncthreads();
    }
    if (threadIdx.x == 0) {
        float m = ssum[0] / rows;
        mean[c] = m;
        var[c]  = ssq[0] / rows - m * m;
    }
}

__global__ void gvae_colstats_f16(const _Float16* __restrict__ src, int rows, int cols,
                                  float* __restrict__ mean, float* __restrict__ var) {
    int c = blockIdx.x;
    __shared__ float ssum[256], ssq[256];
    float s = 0.0f, s2 = 0.0f;
    for (int r = threadIdx.x; r < rows; r += blockDim.x) {
        float v = (float)src[(size_t)r * cols + c];
        s += v; s2 += v * v;
    }
    ssum[threadIdx.x] = s; ssq[threadIdx.x] = s2;
    __syncthreads();
    for (int st = 128; st; st >>= 1) {
        if ((int)threadIdx.x < st) {
            ssum[threadIdx.x] += ssum[threadIdx.x + st];
            ssq[threadIdx.x]  += ssq[threadIdx.x + st];
        }
        __syncthreads();
    }
    if (threadIdx.x == 0) {
        float m = ssum[0] / rows;
        mean[c] = m;
        var[c]  = ssq[0] / rows - m * m;
    }
}

__global__ void gvae_bn_apply(float* __restrict__ h, const float* __restrict__ mean,
                              const float* __restrict__ var, const float* __restrict__ g,
                              const float* __restrict__ b, int total, int cols) {
    int i = blockIdx.x * blockDim.x + threadIdx.x;
    if (i >= total) return;
    int c = i % cols;
    h[i] = g[c] * (h[i] - mean[c]) * rsqrtf(var[c] + 1e-5f) + b[c];
}

__global__ void gvae_make_affine(const float* __restrict__ mean, const float* __restrict__ var,
                                 const float* __restrict__ g, const float* __restrict__ b,
                                 float* __restrict__ s, float* __restrict__ t, int cols) {
    int c = blockIdx.x * blockDim.x + threadIdx.x;
    if (c >= cols) return;
    float sc = g[c] * rsqrtf(var[c] + 1e-5f);
    s[c] = sc;
    t[c] = b[c] - mean[c] * sc;
}

// =======================================================================
// Decoder kernels (WMMA f16 GEMMs)
// =======================================================================
__global__ void gvae_build_feats(const float* __restrict__ z, _Float16* __restrict__ feats) {
    int idx = blockIdx.x * blockDim.x + threadIdx.x;
    if (idx >= RR * 64) return;
    int row = idx >> 6;
    int col = idx & 63;
    int g = row / PP, p = row % PP;
    int i = p / (NPG - 1);
    int jj = p % (NPG - 1);
    int j = jj + (jj >= i ? 1 : 0);
    float val = 0.0f;
    if (col < D_LAT)          val = z[(size_t)(g * NPG + i) * D_LAT + col];
    else if (col < 2 * D_LAT) val = z[(size_t)(g * NPG + j) * D_LAT + (col - D_LAT)];
    feats[idx] = (_Float16)val;
}

// Pack f32 weight [Kin x Ncols] (row-major, k-major) into WMMA B-fragment layout:
// out[(((nt*Kt + kk)*32 + lane)*16 + e], K padded to Kt*32 with zeros.
__global__ void gvae_pack_w(const float* __restrict__ W, int Kin, int Ncols, int Kt,
                            _Float16* __restrict__ out) {
    int idx = blockIdx.x * blockDim.x + threadIdx.x;
    int total = (Ncols >> 4) * Kt * 32 * 16;
    if (idx >= total) return;
    int e    = idx & 15;
    int lane = (idx >> 4) & 31;
    int rest = idx >> 9;
    int kk = rest % Kt;
    int nt = rest / Kt;
    int half = lane >> 4;
    int col  = nt * 16 + (lane & 15);
    int k = kk * 32 + (e < 8 ? half * 8 + e : 16 + half * 8 + (e - 8));
    float val = (k < Kin) ? W[(size_t)k * Ncols + col] : 0.0f;
    out[idx] = (_Float16)val;
}

// C[R x NT*16] (f16) = relu( affine(A) [R x 64] @ Wpacked [64 x NT*16] + bias )
// One wave32 per 16-row stripe; NT accumulators cover ALL column tiles so the
// A fragment is built once per K-step and reused NT times (A traffic = 1 pass).
// K fixed at 64 (Kt = 2, fully unrolled -> 2*NT back-to-back v_wmma).
template <int NT>
__global__ void gvae_gemm_wmma(const _Float16* __restrict__ A,
                               const float* __restrict__ S, const float* __restrict__ T,
                               const _Float16* __restrict__ Bp,
                               const float* __restrict__ bias,
                               _Float16* __restrict__ C,
                               int R) {
    constexpr int K = 64;
    constexpr int Kt = 2;
    constexpr int Ncols = NT * 16;
    int wid  = blockIdx.x * (blockDim.x >> 5) + (threadIdx.x >> 5);
    int lane = threadIdx.x & 31;
    int rowTiles = R >> 4;
    if (wid >= rowTiles) return;
    int mt = wid;
    int rl = lane & 15;
    int half = lane >> 4;
    int row = mt * 16 + rl;     // A row for this lane's fragment

    v8f acc[NT];
#pragma unroll
    for (int nt = 0; nt < NT; ++nt) acc[nt] = v8f{};

    const _Float16* Arow = A + (size_t)row * K;
#pragma unroll
    for (int kk = 0; kk < Kt; ++kk) {
        int kb1 = kk * 32 + half * 8;
        int kb2 = kb1 + 16;
        v16h a;
        if (S) {
#pragma unroll
            for (int e = 0; e < 8; ++e) {
                int k1 = kb1 + e, k2 = kb2 + e;
                a[e]     = (_Float16)((float)Arow[k1] * S[k1] + T[k1]);
                a[e + 8] = (_Float16)((float)Arow[k2] * S[k2] + T[k2]);
            }
        } else {
            v8h alo = *(const v8h*)(Arow + kb1);
            v8h ahi = *(const v8h*)(Arow + kb2);
#pragma unroll
            for (int e = 0; e < 8; ++e) { a[e] = alo[e]; a[e + 8] = ahi[e]; }
        }
#pragma unroll
        for (int nt = 0; nt < NT; ++nt) {
            v16h b = *(const v16h*)(Bp + ((((size_t)nt * Kt + kk) * 32 + lane) << 4));
            acc[nt] = __builtin_amdgcn_wmma_f32_16x16x32_f16(false, a, false, b,
                                                             (short)0, acc[nt], false, false);
        }
    }
#pragma unroll
    for (int nt = 0; nt < NT; ++nt) {
        int col = nt * 16 + rl;
        float bcol = bias[col];
#pragma unroll
        for (int i = 0; i < 8; ++i) {
            int r = mt * 16 + i + half * 8;
            float v = acc[nt][i] + bcol;
            v = v > 0.0f ? v : 0.0f;
            C[(size_t)r * Ncols + col] = (_Float16)v;
        }
    }
}

// out[row] = affine(x128[row,:]) . Wf + bf      (wave32 per row)
__global__ void gvae_dense_out(const _Float16* __restrict__ x128,
                               const float* __restrict__ S, const float* __restrict__ T,
                               const float* __restrict__ Wf, const float* __restrict__ bf,
                               float* __restrict__ out, int rows) {
    int wid  = blockIdx.x * (blockDim.x >> 5) + (threadIdx.x >> 5);
    int lane = threadIdx.x & 31;
    if (wid >= rows) return;
    const _Float16* r = x128 + (size_t)wid * 128;
    float a = 0.0f;
#pragma unroll
    for (int c0 = 0; c0 < 128; c0 += 32) {
        int c = c0 + lane;
        float v = (float)r[c] * S[c] + T[c];
        a += v * Wf[c];
    }
    for (int off = 16; off; off >>= 1) a += __shfl_down(a, off, 32);
    if (lane == 0) out[wid] = a + bf[0];
}

// =======================================================================
// Host driver
// =======================================================================
extern "C" void kernel_launch(void* const* d_in, const int* in_sizes, int n_in,
                              void* d_out, int out_size, void* d_ws, size_t ws_size,
                              hipStream_t stream) {
    (void)in_sizes; (void)n_in; (void)out_size; (void)ws_size;
    auto F = [&](int i) { return (const float*)d_in[i]; };
    const float* x  = F(0);
    const float* ea = F(1);
    const int*   ei = (const int*)d_in[2];
    float* out = (float*)d_out;

    // ---------------- workspace carve-out ----------------
    char* ws = (char*)d_ws;
    size_t off = 0;
    auto alloc = [&](size_t bytes) -> char* {
        char* p = ws + off;
        off = (off + bytes + 255) & ~(size_t)255;
        return p;
    };
    float* qb   = (float*)alloc((size_t)NN * 141 * 4);
    float* kb   = (float*)alloc((size_t)NN * 141 * 4);
    float* vb   = (float*)alloc((size_t)NN * 141 * 4);
    float* xrb  = (float*)alloc((size_t)NN * 47 * 4);
    float* accb = (float*)alloc((size_t)NN * 141 * 4);
    float* alph = (float*)alloc((size_t)MM * HEADS * 4);
    float* amax = (float*)alloc((size_t)NN * HEADS * 4);
    float* denb = (float*)alloc((size_t)NN * HEADS * 4);
    float* hA   = (float*)alloc((size_t)NN * 47 * 4);
    float* hB   = (float*)alloc((size_t)NN * 47 * 4);
    float* zb   = (float*)alloc((size_t)NN * D_LAT * 4);
    float* meanb = (float*)alloc(512);
    float* varb  = (float*)alloc(512);
    float* sa1 = (float*)alloc(512); float* ta1 = (float*)alloc(512);
    float* sa2 = (float*)alloc(512); float* ta2 = (float*)alloc(512);
    float* sa3 = (float*)alloc(512); float* ta3 = (float*)alloc(512);
    _Float16* wpack = (_Float16*)alloc(16384);
    _Float16* feats = (_Float16*)alloc((size_t)RR * 64 * 2);
    _Float16* h1    = (_Float16*)alloc((size_t)RR * 64 * 2);
    _Float16* h2    = (_Float16*)alloc((size_t)RR * 64 * 2);
    _Float16* x128  = (_Float16*)alloc((size_t)RR * 128 * 2);

    auto cdiv = [](int a, int b) { return (a + b - 1) / b; };

    // ---------------- encoder: 4 TransformerConv layers ----------------
    struct L { int din, dout, base, bng; bool relu_bn; const float* in; float* outp; };
    L layers[4] = {
        {F_IN, 47, 4,  14, true,  x,  hA},
        {47,   36, 16, 26, true,  hA, hB},
        {36,   36, 28, 38, true,  hB, hA},
        {36, D_LAT, 40, -1, false, hA, zb},
    };
    for (int li = 0; li < 4; ++li) {
        const L& lc = layers[li];
        int hd = HEADS * lc.dout;
        gvae_qkv<<<cdiv(NN * hd, 256), 256, 0, stream>>>(
            lc.in, NN, lc.din, hd, lc.dout,
            F(lc.base + 0), F(lc.base + 1), F(lc.base + 2), F(lc.base + 3),
            F(lc.base + 4), F(lc.base + 5), F(lc.base + 7), F(lc.base + 8),
            qb, kb, vb, xrb);
        gvae_attn_init<<<cdiv(NN * hd, 256), 256, 0, stream>>>(
            amax, denb, accb, NN * HEADS, NN * hd);
        gvae_alpha<<<cdiv(MM * HEADS, 256), 256, 0, stream>>>(
            qb, kb, ea, F(lc.base + 6), ei, MM, lc.dout, hd, alph, amax);
        gvae_exden<<<cdiv(MM * HEADS, 256), 256, 0, stream>>>(
            ei, MM, amax, alph, denb);
        gvae_scatter<<<cdiv(MM * HEADS, 256), 256, 0, stream>>>(
            vb, ea, F(lc.base + 6), ei, MM, lc.dout, hd, alph, denb, accb);
        gvae_combine<<<cdiv(NN, 256), 256, 0, stream>>>(
            accb, xrb, F(lc.base + 9), NN, lc.dout, hd, lc.relu_bn ? 1 : 0, lc.outp);
        if (lc.relu_bn) {
            gvae_colstats_f32<<<lc.dout, 256, 0, stream>>>(lc.outp, NN, lc.dout, meanb, varb);
            gvae_bn_apply<<<cdiv(NN * lc.dout, 256), 256, 0, stream>>>(
                lc.outp, meanb, varb, F(lc.bng), F(lc.bng + 1), NN * lc.dout, lc.dout);
        }
    }

    // ---------------- decoder: WMMA GEMM stack ----------------
    gvae_build_feats<<<cdiv(RR * 64, 256), 256, 0, stream>>>(zb, feats);

    const int rowTiles = RR / 16;                       // 36480
    const int gemmBlocks = rowTiles / 8;                // 8 waves per block

    // dec1: [R,48->64pad] @ [64,64]  (no BN on input)
    gvae_pack_w<<<cdiv(4 * 2 * 512, 256), 256, 0, stream>>>(F(50), 48, 64, 2, wpack);
    gvae_gemm_wmma<4><<<gemmBlocks, 256, 0, stream>>>(
        feats, nullptr, nullptr, wpack, F(51), h1, RR);
    gvae_colstats_f16<<<64, 256, 0, stream>>>(h1, RR, 64, meanb, varb);
    gvae_make_affine<<<1, 128, 0, stream>>>(meanb, varb, F(52), F(53), sa1, ta1, 64);

    // dec2: BN1(h1) @ [64,64]
    gvae_pack_w<<<cdiv(4 * 2 * 512, 256), 256, 0, stream>>>(F(54), 64, 64, 2, wpack);
    gvae_gemm_wmma<4><<<gemmBlocks, 256, 0, stream>>>(
        h1, sa1, ta1, wpack, F(55), h2, RR);
    gvae_colstats_f16<<<64, 256, 0, stream>>>(h2, RR, 64, meanb, varb);
    gvae_make_affine<<<1, 128, 0, stream>>>(meanb, varb, F(56), F(57), sa2, ta2, 64);

    // three heads from d = BN2(h2): (adj, logits1, logits2) in output order
    const int headW[3]  = {66, 58, 62};   // d2_adj, dec22, d3
    const int headBN[3] = {68, 60, 64};   // dbn2_adj, dbn22, dbn3
    const int headD[3]  = {74, 70, 72};   // dense_adj, dense4, dense5
    float* outp[3] = {out, out + RR, out + 2 * (size_t)RR};
    for (int t = 0; t < 3; ++t) {
        gvae_pack_w<<<cdiv(8 * 2 * 512, 256), 256, 0, stream>>>(F(headW[t]), 64, 128, 2, wpack);
        gvae_gemm_wmma<8><<<gemmBlocks, 256, 0, stream>>>(
            h2, sa2, ta2, wpack, F(headW[t] + 1), x128, RR);
        gvae_colstats_f16<<<128, 256, 0, stream>>>(x128, RR, 128, meanb, varb);
        gvae_make_affine<<<1, 128, 0, stream>>>(meanb, varb, F(headBN[t]), F(headBN[t] + 1),
                                                sa3, ta3, 128);
        gvae_dense_out<<<RR / 8, 256, 0, stream>>>(x128, sa3, ta3,
                                                   F(headD[t]), F(headD[t] + 1), outp[t], RR);
    }
}
